// ShiftedWindowAttention_65755949302020
// MI455X (gfx1250) — matrix operand order repaired and checked by reading, not compiled, not run on hardware
//
#include <hip/hip_runtime.h>
#include <hip/hip_bf16.h>
#include <math.h>

typedef __attribute__((ext_vector_type(16))) _Float16 v16h;
typedef __attribute__((ext_vector_type(8)))  _Float16 v8h;
typedef __attribute__((ext_vector_type(4)))  _Float16 v4h;
typedef __attribute__((ext_vector_type(2)))  _Float16 v2h;
typedef __attribute__((ext_vector_type(8)))  float    v8f;

// ---------------- problem constants ----------------
constexpr int Bn   = 2;
constexpr int H    = 2048;
constexpr int W    = 32;
constexpr int C    = 256;
constexpr int NH   = 8;
constexpr int HD   = 32;      // head dim
constexpr int WH   = 16;
constexpr int WW   = 8;
constexpr int SHR  = 8;       // shift h
constexpr int SWR  = 4;       // shift w
constexpr int N    = WH * WW; // 128 tokens / window
constexpr int NHW  = H / WH;  // 128
constexpr int NWW  = W / WW;  // 4
constexpr int NWIN = Bn * NHW * NWW; // 1024 windows
constexpr int CPB  = 512;

// ---------------- LDS layout (fused kernel, dynamic) ----------------
constexpr int SX_STRIDE  = C + 8;     // 264 halfs
constexpr int SP_STRIDE  = N + 8;     // 136 halfs
constexpr int SQ_STRIDE  = HD + 8;    // 40 halfs
constexpr int SX_BYTES   = N * SX_STRIDE * 2;       // 67584
constexpr int SCTX_OFF   = SX_BYTES;                // 67584
constexpr int SP_OFF     = SCTX_OFF + SX_BYTES;     // 135168
constexpr int SQ_OFF     = SP_OFF + N * SP_STRIDE * 2;   // 169984
constexpr int SK_OFF     = SQ_OFF + N * SQ_STRIDE * 2;   // 180224
constexpr int SVT_OFF    = SK_OFF + N * SQ_STRIDE * 2;   // 190464
constexpr int SMEM_BYTES = SVT_OFF + HD * SP_STRIDE * 2; // 199168

// ---------------- fragment helpers (contiguous k order, same for A and B) ----------------
// A-matrix 16x32 f16: lane = halfid*16 + (m - m0); element e -> VGPR r=e/2, t=e%2
//   K(e) = 16*halfid + 2*r + t
__device__ inline v16h load_a_frag(const _Float16* __restrict__ row, int kbase, int halfid) {
  v16h a;
#pragma unroll
  for (int r = 0; r < 8; ++r) {
    int k = kbase + 16 * halfid + 2 * r;
    v2h t = *(const v2h*)(row + k);
    a[2 * r]     = t[0];
    a[2 * r + 1] = t[1];
  }
  return a;
}

// B-matrix 32x16 f16: lane holds column n = l%16; element e -> K = 16*(lane/16) + e
// i.e. 16 contiguous K values -> one 32B read per lane (here two 16B reads).
__device__ inline v16h load_b_frag16(const _Float16* __restrict__ p /*16B aligned*/) {
  v8h lo = *(const v8h*)p;
  v8h hi = *(const v8h*)(p + 8);
  v16h r;
#pragma unroll
  for (int e = 0; e < 8; ++e) { r[e] = lo[e]; r[e + 8] = hi[e]; }
  return r;
}

__device__ inline v8f zero_acc() {
  v8f z = {0.f, 0.f, 0.f, 0.f, 0.f, 0.f, 0.f, 0.f};
  return z;
}

__device__ inline v8f wmma_f16(v16h a, v16h b, v8f c) {
  return __builtin_amdgcn_wmma_f32_16x16x32_f16(false, a, false, b, (short)0, c,
                                                false, false);
}

// shifted-window region id in rolled coordinate space
__device__ inline int region_id(int whi, int wwi, int tok) {
  int row = whi * WH + (tok >> 3);
  int col = wwi * WW + (tok & 7);
  int hid = (row >= H - WH) + (row >= H - SHR);
  int wid = (col >= W - WW) + (col >= W - SWR);
  return hid * 3 + wid;
}

// ---------------- kernel 0: fp32 -> f16 weight convert ----------------
__global__ __launch_bounds__(256) void convert_weights_kernel(
    const float* __restrict__ qkv_w, const float* __restrict__ proj_w,
    _Float16* __restrict__ qkv_h, _Float16* __restrict__ proj_h) {
  int idx = blockIdx.x * 256 + threadIdx.x;
  if (idx < 3 * C * C) qkv_h[idx] = (_Float16)qkv_w[idx];
  if (idx < C * C)     proj_h[idx] = (_Float16)proj_w[idx];
}

// ---------------- kernel 1: continuous relative position bias ----------------
// bias[b][h][i][j] = sum_k relu(rel . w1[k] + b1[k]) * w2[h][k]
__global__ __launch_bounds__(256) void cpb_bias_kernel(
    const float* __restrict__ loc, const float* __restrict__ w1,
    const float* __restrict__ b1, const float* __restrict__ w2,
    float* __restrict__ bias) {
  __shared__ float s_w1[CPB * 3];
  __shared__ float s_b1[CPB];
  __shared__ float s_w2[NH * CPB];
  __shared__ float s_loc[WW * 2];

  int t = threadIdx.x;
  int b = blockIdx.x >> 6;            // 64 blocks per batch
  for (int i = t; i < CPB * 3; i += 256) s_w1[i] = w1[i];
  for (int i = t; i < CPB; i += 256)     s_b1[i] = b1[i];
  for (int i = t; i < NH * CPB; i += 256) s_w2[i] = w2[i];
  if (t < WW * 2) s_loc[t] = loc[b * WW * 2 + t];
  __syncthreads();

  int pair = ((blockIdx.x & 63) << 8) + t;   // 0..16383
  int i = pair >> 7, j = pair & 127;
  float d0 = (float)(i >> 3) - (float)(j >> 3);
  float d1 = s_loc[(i & 7) * 2]     - s_loc[(j & 7) * 2];
  float d2 = s_loc[(i & 7) * 2 + 1] - s_loc[(j & 7) * 2 + 1];

  float acc[NH];
#pragma unroll
  for (int h = 0; h < NH; ++h) acc[h] = 0.f;

  for (int k = 0; k < CPB; ++k) {
    float hv = fmaf(d0, s_w1[k * 3], fmaf(d1, s_w1[k * 3 + 1],
               fmaf(d2, s_w1[k * 3 + 2], s_b1[k])));
    hv = fmaxf(hv, 0.f);
#pragma unroll
    for (int h = 0; h < NH; ++h) acc[h] = fmaf(hv, s_w2[h * CPB + k], acc[h]);
  }
#pragma unroll
  for (int h = 0; h < NH; ++h)
    bias[(((size_t)b * NH + h) * N + i) * N + j] = acc[h];
}

// ---------------- kernel 2: fused shifted-window attention ----------------
// One workgroup (8 wave32) per window. 1024 windows.
__global__ __launch_bounds__(256) void swin_fused_kernel(
    const float* __restrict__ x, const _Float16* __restrict__ qkv_h,
    const _Float16* __restrict__ proj_h, const float* __restrict__ qkv_b,
    const float* __restrict__ proj_b, const float* __restrict__ bias,
    float* __restrict__ out) {
  extern __shared__ char smem[];
  _Float16* sX   = (_Float16*)smem;               // [N][SX_STRIDE] window input
  _Float16* sCtx = (_Float16*)(smem + SCTX_OFF);  // [N][SX_STRIDE] attn output
  _Float16* sP   = (_Float16*)(smem + SP_OFF);    // [N][SP_STRIDE] probabilities
  _Float16* sQ   = (_Float16*)(smem + SQ_OFF);    // [N][SQ_STRIDE]
  _Float16* sK   = (_Float16*)(smem + SK_OFF);    // [N][SQ_STRIDE]
  _Float16* sVt  = (_Float16*)(smem + SVT_OFF);   // [HD][SP_STRIDE] V transposed

  const int tid    = threadIdx.x;
  const int wave   = tid >> 5;
  const int lane   = tid & 31;
  const int halfid = lane >> 4;
  const int l16    = lane & 15;
  const int m0     = wave * 16;            // this wave's M strip

  const int win  = blockIdx.x;
  const int b    = win >> 9;
  const int widx = win & 511;
  const int whi  = widx >> 2;              // window row (nHw = 128)
  const int wwi  = widx & 3;               // window col (nWw = 4)
  const float scale = 0.17677669529663687f; // 32^-0.5

  // ---- phase A: gather rolled window -> LDS f16 ----
  {
    // 128 tokens x 256 ch = 8192 float4 loads
#pragma unroll 4
    for (int it = 0; it < 32; ++it) {
      int v = tid + it * 256;
      int tok = v >> 6;
      int cp  = (v & 63) * 4;
      int gh = (whi * WH + (tok >> 3) + SHR) & (H - 1);
      int gw = (wwi * WW + (tok & 7) + SWR) & (W - 1);
      const float4 f = *(const float4*)(x + (((size_t)b * H + gh) * W + gw) * C + cp);
      v4h hv;
      hv[0] = (_Float16)f.x; hv[1] = (_Float16)f.y;
      hv[2] = (_Float16)f.z; hv[3] = (_Float16)f.w;
      *(v4h*)(sX + tok * SX_STRIDE + cp) = hv;
    }
  }

  const int my_reg_half0 = 0; (void)my_reg_half0;

  // ---- per-head loop ----
  for (int h = 0; h < NH; ++h) {
    __syncthreads();  // sX ready / previous head done with sQ,sK,sVt,sP

    // ---- QKV GEMM for this head: M=128, K=256, N=96 (q0 q1 k0 k1 v0 v1) ----
    int colt[6];
    colt[0] = h * HD;           colt[1] = h * HD + 16;
    colt[2] = C + h * HD;       colt[3] = C + h * HD + 16;
    colt[4] = 2 * C + h * HD;   colt[5] = 2 * C + h * HD + 16;

    v8f acc[6];
#pragma unroll
    for (int t = 0; t < 6; ++t) acc[t] = zero_acc();

    const _Float16* arow = sX + (m0 + l16) * SX_STRIDE;
#pragma unroll
    for (int ks = 0; ks < C / 32; ++ks) {
      v16h a = load_a_frag(arow, ks * 32, halfid);
#pragma unroll
      for (int t = 0; t < 6; ++t) {
        const _Float16* bp = qkv_h + (size_t)(colt[t] + l16) * C + ks * 32 + 16 * halfid;
        acc[t] = wmma_f16(a, load_b_frag16(bp), acc[t]);
      }
    }
    // bias add + scatter q/k/v to LDS
#pragma unroll
    for (int t = 0; t < 6; ++t) {
      float qb = qkv_b[colt[t] + l16];
#pragma unroll
      for (int r = 0; r < 8; ++r) {
        int m = m0 + 8 * halfid + r;
        float val = acc[t][r] + qb;
        if (t < 2) {
          sQ[m * SQ_STRIDE + (t * 16 + l16)] = (_Float16)(val * scale);
        } else if (t < 4) {
          sK[m * SQ_STRIDE + ((t - 2) * 16 + l16)] = (_Float16)val;
        } else {
          sVt[((t - 4) * 16 + l16) * SP_STRIDE + m] = (_Float16)val;
        }
      }
    }
    __syncthreads();  // q,k,v ready

    // ---- S = Q @ K^T  (M=128, N=128, K=32) + bias + mask, softmax ----
    v8f s[8];
    {
      v16h a = load_a_frag(sQ + (m0 + l16) * SQ_STRIDE, 0, halfid);
#pragma unroll
      for (int nt = 0; nt < 8; ++nt) {
        const _Float16* bp = sK + (nt * 16 + l16) * SQ_STRIDE + 16 * halfid;
        s[nt] = wmma_f16(a, load_b_frag16(bp), zero_acc());
      }
    }
    {
      const float* bp = bias + (((size_t)b * NH + h) * N) * N;
#pragma unroll
      for (int nt = 0; nt < 8; ++nt) {
        int j = nt * 16 + l16;
        int rj = region_id(whi, wwi, j);
#pragma unroll
        for (int r = 0; r < 8; ++r) {
          int i = m0 + 8 * halfid + r;
          float add = bp[i * N + j];
          if (region_id(whi, wwi, i) != rj) add -= 100.0f;
          s[nt][r] += add;
        }
      }
    }
    // row softmax: each lane owns 8 rows; reduce across 16 lanes of its half
#pragma unroll
    for (int r = 0; r < 8; ++r) {
      float mx = -3.4e38f;
#pragma unroll
      for (int nt = 0; nt < 8; ++nt) mx = fmaxf(mx, s[nt][r]);
#pragma unroll
      for (int mk = 1; mk < 16; mk <<= 1) mx = fmaxf(mx, __shfl_xor(mx, mk));
      float sum = 0.f;
#pragma unroll
      for (int nt = 0; nt < 8; ++nt) {
        float e = __expf(s[nt][r] - mx);
        s[nt][r] = e;
        sum += e;
      }
#pragma unroll
      for (int mk = 1; mk < 16; mk <<= 1) sum += __shfl_xor(sum, mk);
      float inv = __frcp_rn(sum);
      int i = m0 + 8 * halfid + r;
#pragma unroll
      for (int nt = 0; nt < 8; ++nt)
        sP[i * SP_STRIDE + nt * 16 + l16] = (_Float16)(s[nt][r] * inv);
    }
    __syncthreads();  // P ready

    // ---- O = P @ V  (M=128, K=128, N=32) ----
    v8f o0 = zero_acc(), o1 = zero_acc();
    const _Float16* prow = sP + (m0 + l16) * SP_STRIDE;
#pragma unroll
    for (int ks = 0; ks < N / 32; ++ks) {
      v16h a = load_a_frag(prow, ks * 32, halfid);
      const _Float16* bp0 = sVt + (l16) * SP_STRIDE + ks * 32 + 16 * halfid;
      const _Float16* bp1 = sVt + (16 + l16) * SP_STRIDE + ks * 32 + 16 * halfid;
      o0 = wmma_f16(a, load_b_frag16(bp0), o0);
      o1 = wmma_f16(a, load_b_frag16(bp1), o1);
    }
#pragma unroll
    for (int r = 0; r < 8; ++r) {
      int m = m0 + 8 * halfid + r;
      sCtx[m * SX_STRIDE + h * HD + l16]      = (_Float16)o0[r];
      sCtx[m * SX_STRIDE + h * HD + 16 + l16] = (_Float16)o1[r];
    }
  }
  __syncthreads();  // ctx ready

  // ---- output projection: M=128, K=256, N=256 (two passes of 8 N-tiles) ----
  // fused reverse window partition + roll(+sh,+sw) scatter
  size_t rowbase[8];
#pragma unroll
  for (int r = 0; r < 8; ++r) {
    int m = m0 + 8 * halfid + r;
    int fh = (whi * WH + (m >> 3) + SHR) & (H - 1);
    int fw = (wwi * WW + (m & 7) + SWR) & (W - 1);
    rowbase[r] = (((size_t)b * H + fh) * W + fw) * C;
  }
  const _Float16* crow = sCtx + (m0 + l16) * SX_STRIDE;
#pragma unroll
  for (int np = 0; np < 2; ++np) {
    v8f accp[8];
#pragma unroll
    for (int nt = 0; nt < 8; ++nt) accp[nt] = zero_acc();
#pragma unroll
    for (int ks = 0; ks < C / 32; ++ks) {
      v16h a = load_a_frag(crow, ks * 32, halfid);
#pragma unroll
      for (int nt = 0; nt < 8; ++nt) {
        int gcol = np * 128 + nt * 16 + l16;
        const _Float16* bp = proj_h + (size_t)gcol * C + ks * 32 + 16 * halfid;
        accp[nt] = wmma_f16(a, load_b_frag16(bp), accp[nt]);
      }
    }
#pragma unroll
    for (int nt = 0; nt < 8; ++nt) {
      int n = np * 128 + nt * 16 + l16;
      float pb = proj_b[n];
#pragma unroll
      for (int r = 0; r < 8; ++r)
        out[rowbase[r] + n] = accp[nt][r] + pb;
    }
  }
}

// ---------------- host launcher ----------------
extern "C" void kernel_launch(void* const* d_in, const int* in_sizes, int n_in,
                              void* d_out, int out_size, void* d_ws, size_t ws_size,
                              hipStream_t stream) {
  const float* x      = (const float*)d_in[0];
  const float* loc    = (const float*)d_in[1];
  const float* qkv_w  = (const float*)d_in[2];
  const float* qkv_b  = (const float*)d_in[3];
  const float* proj_w = (const float*)d_in[4];
  const float* proj_b = (const float*)d_in[5];
  const float* cpb_w1 = (const float*)d_in[6];
  const float* cpb_b1 = (const float*)d_in[7];
  const float* cpb_w2 = (const float*)d_in[8];
  float* out = (float*)d_out;

  // workspace layout: qkv_w f16 | proj_w f16 | bias f32
  _Float16* qkv_h  = (_Float16*)d_ws;
  _Float16* proj_h = qkv_h + 3 * C * C;
  float*    bias   = (float*)((char*)d_ws + (size_t)(3 * C * C + C * C) * sizeof(_Float16));

  convert_weights_kernel<<<(3 * C * C + 255) / 256, 256, 0, stream>>>(
      qkv_w, proj_w, qkv_h, proj_h);

  cpb_bias_kernel<<<Bn * (N * N / 256), 256, 0, stream>>>(
      loc, cpb_w1, cpb_b1, cpb_w2, bias);

  (void)hipFuncSetAttribute((const void*)swin_fused_kernel,
                            hipFuncAttributeMaxDynamicSharedMemorySize, SMEM_BYTES);
  swin_fused_kernel<<<NWIN, 256, SMEM_BYTES, stream>>>(
      x, qkv_h, proj_h, qkv_b, proj_b, bias, out);
}
